// Atom91Head_67405216743688
// MI455X (gfx1250) — matrix operand order repaired and checked by reading, not compiled, not run on hardware
//
#include <hip/hip_runtime.h>
#include <hip/hip_bf16.h>
#include <stdint.h>

// ---------------- problem constants ----------------
#define NN      4096
#define EE      32768
#define D0      32
#define D1      16
#define AAC     20
#define COUTC   91
#define EDGED   17
#define MIDC    32
#define NORM_CLAMP 5.9604644775390625e-8f   // 2^-24
#define EPSV    1e-5f

// GEMM structure: out[(e,m),k] = sum_q X[(e,m),q] * Wbig[k,q]
//   q < 1024        : j = q>>5,           p = q&31            (deg-0 path, p = c)
//   q >= 1024       : j = 32+(q-1024)/48, p = 32+(q-1024)%48  (deg-1 path, p = 32+c*3+f)
#define KTOT    2560          // 1024 + 1536
#define KSTEPS  80            // KTOT / 32
#define NT      6             // 96 = 6*16 output cols (91 padded to 96)
#define CHUNK   256           // K chunk built in LDS at a time
#define NCHUNK  10
#define KSPC    8             // k-steps per chunk (CHUNK/32)
#define XSTRIDE 264           // LDS row stride in halves (pad to dodge bank conflicts)
#define ETILES  (EE/16)       // 2048

typedef __attribute__((ext_vector_type(16))) _Float16 v16h;
typedef __attribute__((ext_vector_type(8)))  _Float16 v8h;
typedef __attribute__((ext_vector_type(2)))  _Float16 h2;
typedef __attribute__((ext_vector_type(8)))  float    v8f;

// ---------------------------------------------------------------------------
// Kernel P1: node transition LinearSE3 + NormSE3  -> t0f [N,32], t1f [N,16,3]
// ---------------------------------------------------------------------------
__global__ __launch_bounds__(64) void node_prep_kernel(
    const float* __restrict__ d0, const float* __restrict__ d1,
    const float* __restrict__ s0,
    const float* __restrict__ Wt0, const float* __restrict__ Wt1,
    const float* __restrict__ ln0w, const float* __restrict__ ln0b,
    const float* __restrict__ ln1w, const float* __restrict__ ln1b,
    float* __restrict__ t0f, float* __restrict__ t1f)
{
    const int n = blockIdx.x;
    const int t = threadIdx.x;
    if (t < 32) {
        const int c = t;
        float v = 0.f;
        #pragma unroll
        for (int i = 0; i < 32; ++i) v += Wt0[c*52 + i] * d0[n*32 + i];
        #pragma unroll
        for (int i = 0; i < 20; ++i) v += Wt0[c*52 + 32 + i] * s0[n*20 + i];
        float nk = fmaxf(fabsf(v), NORM_CLAMP);
        float s = nk;
        #pragma unroll
        for (int m = 1; m < 32; m <<= 1) s += __shfl_xor(s, m, 32);
        const float mu = s * (1.f/32.f);
        const float dn = nk - mu;
        float vv = dn*dn;
        #pragma unroll
        for (int m = 1; m < 32; m <<= 1) vv += __shfl_xor(vv, m, 32);
        const float var = vv * (1.f/32.f);
        const float nn = fmaxf(dn * rsqrtf(var + EPSV) * ln0w[c] + ln0b[c], 0.f);
        t0f[n*32 + c] = v / nk * nn;
    } else if (t < 48) {
        const int c = t - 32;
        float y0 = 0.f, y1 = 0.f, y2 = 0.f;
        #pragma unroll
        for (int i = 0; i < 16; ++i) {
            const float w = Wt1[c*16 + i];
            y0 += w * d1[n*48 + i*3 + 0];
            y1 += w * d1[n*48 + i*3 + 1];
            y2 += w * d1[n*48 + i*3 + 2];
        }
        float nk = fmaxf(sqrtf(y0*y0 + y1*y1 + y2*y2), NORM_CLAMP);
        float s = nk;
        #pragma unroll
        for (int m = 1; m < 16; m <<= 1) s += __shfl_xor(s, m, 16);
        const float mu = s * (1.f/16.f);
        const float dn = nk - mu;
        float vv = dn*dn;
        #pragma unroll
        for (int m = 1; m < 16; m <<= 1) vv += __shfl_xor(vv, m, 16);
        const float var = vv * (1.f/16.f);
        const float nn = fmaxf(dn * rsqrtf(var + EPSV) * ln1w[c] + ln1b[c], 0.f);
        const float sc = nn / nk;
        t1f[n*48 + c*3 + 0] = y0 * sc;
        t1f[n*48 + c*3 + 1] = y1 * sc;
        t1f[n*48 + c*3 + 2] = y2 * sc;
    }
}

// ---------------------------------------------------------------------------
// Kernel P2: pack Wbig (fp16) in per-lane B order (contiguous k, matching the A loads).
// Element (nt, gk, lane, t):  k = nt*16 + lane%16 ; q = gk*32 + (lane/16)*16 + t
// ---------------------------------------------------------------------------
__global__ __launch_bounds__(256) void pack_wbig_kernel(
    const float* __restrict__ W3a,   // r01_W3 [91*32, 32]
    const float* __restrict__ W3b,   // r11_W3 [91*48, 32]
    _Float16* __restrict__ Bpack)
{
    const int idx = blockIdx.x * 256 + threadIdx.x;   // < NT*KSTEPS*32*16
    const int tt   = idx & 15;
    int rest = idx >> 4;
    const int lane = rest & 31;  rest >>= 5;
    const int gk   = rest % KSTEPS;
    const int nt   = rest / KSTEPS;
    const int k = nt*16 + (lane & 15);
    const int q = gk*32 + (lane >> 4)*16 + tt;
    float val = 0.f;
    if (k < COUTC) {
        if (q < 1024) {
            const int j = q >> 5, c = q & 31;
            val = W3a[(k*32 + c)*32 + j];
        } else {
            const int qq = q - 1024;
            const int j = qq / 48, cf = qq % 48;
            val = W3b[(k*48 + cf)*32 + j];
        }
    }
    Bpack[idx] = (_Float16)val;
}

// ---------------------------------------------------------------------------
// Kernel P3: per-edge radial MLP hiddens h[E,64] and z[E,80,3] (fp16)
// ---------------------------------------------------------------------------
__global__ __launch_bounds__(32) void edge_prep_kernel(
    const float* __restrict__ es, const float* __restrict__ b01,
    const float* __restrict__ b11,
    const float* __restrict__ W1a, const float* __restrict__ b1a,
    const float* __restrict__ W2a, const float* __restrict__ b2a,
    const float* __restrict__ W1b, const float* __restrict__ b1b,
    const float* __restrict__ W2b, const float* __restrict__ b2b,
    const int* __restrict__ src,
    const float* __restrict__ t0f, const float* __restrict__ t1f,
    _Float16* __restrict__ Hws, _Float16* __restrict__ Zws)
{
    __shared__ float esl[EDGED];
    __shared__ float h1[32];
    const int e = blockIdx.x;
    const int j = threadIdx.x;
    if (j < EDGED) esl[j] = es[e*EDGED + j];
    __syncthreads();

    // ---- path (0,1) MLP ----
    float a = b1a[j];
    #pragma unroll
    for (int l = 0; l < EDGED; ++l) a += W1a[j*EDGED + l] * esl[l];
    h1[j] = fmaxf(a, 0.f);
    __syncthreads();
    float h = b2a[j];
    #pragma unroll
    for (int l = 0; l < 32; ++l) h += W2a[j*32 + l] * h1[l];
    Hws[(size_t)e*64 + j] = (_Float16)fmaxf(h, 0.f);
    __syncthreads();

    // ---- path (1,1) MLP ----
    a = b1b[j];
    #pragma unroll
    for (int l = 0; l < EDGED; ++l) a += W1b[j*EDGED + l] * esl[l];
    h1[j] = fmaxf(a, 0.f);
    __syncthreads();
    h = b2b[j];
    #pragma unroll
    for (int l = 0; l < 32; ++l) h += W2b[j*32 + l] * h1[l];
    Hws[(size_t)e*64 + 32 + j] = (_Float16)fmaxf(h, 0.f);

    // ---- z = gathered feature x basis ----
    const int s = src[e];
    for (int idx = j; idx < 96; idx += 32) {            // deg-0: z[c][m] = t0*b01
        const int c = idx / 3, m = idx % 3;
        Zws[(size_t)e*240 + c*3 + m] = (_Float16)(t0f[s*32 + c] * b01[e*3 + m]);
    }
    for (int idx = j; idx < 144; idx += 32) {           // deg-1: z[32+cf][m]
        const int cf = idx / 3, m = idx % 3;
        const int c = cf / 3, f = cf % 3;
        float v = 0.f;
        #pragma unroll
        for (int i = 0; i < 3; ++i)
            v += t1f[s*48 + c*3 + i] * b11[e*27 + i*9 + f*3 + m];
        Zws[(size_t)e*240 + (32 + cf)*3 + m] = (_Float16)v;
    }
}

// ---------------------------------------------------------------------------
// Kernel C: fused radial-W3 x CG-contraction GEMM via WMMA + atomic scatter.
// One 32-thread wave per tile of 16 edges; ALL THREE m components are
// accumulated by the same wave so each B fetch feeds 3 WMMAs.
// M=16 edges, N=96 k, K=2560, 3 accumulator sets (18 x v8f).
// ---------------------------------------------------------------------------
__global__ __launch_bounds__(32) void conv_gemm_kernel(
    const _Float16* __restrict__ Hws, const _Float16* __restrict__ Zws,
    const _Float16* __restrict__ Bpack, const int* __restrict__ dst,
    float* __restrict__ atoms)
{
    __shared__ _Float16 hs[16][64];            // 2 KB
    __shared__ _Float16 zs[3][16][80];         // 7.5 KB  (z transposed: [m][row][p])
    __shared__ _Float16 Xs[3][16][XSTRIDE];    // 25.3 KB
    __shared__ int      dsts[16];

    const int et    = blockIdx.x;
    const int ebase = et * 16;
    const int lane  = threadIdx.x;
    const int g     = lane >> 4;       // 0/1
    const int row   = lane & 15;       // A-operand M row for this lane

    for (int idx = lane; idx < 16*64; idx += 32) {
        const int r = idx >> 6, jj = idx & 63;
        hs[r][jj] = Hws[(size_t)(ebase + r)*64 + jj];
    }
    for (int idx = lane; idx < 16*240; idx += 32) {
        const int r = idx / 240, pm = idx % 240;
        const int p = pm / 3, m = pm % 3;
        zs[m][r][p] = Zws[(size_t)(ebase + r)*240 + pm];
    }
    if (lane < 16) dsts[lane] = dst[ebase + lane];
    __syncthreads();

    v8f acc[3][NT];
    #pragma unroll
    for (int m = 0; m < 3; ++m)
        #pragma unroll
        for (int nt = 0; nt < NT; ++nt)
            #pragma unroll
            for (int r = 0; r < 8; ++r) acc[m][nt][r] = 0.f;

    for (int ch = 0; ch < NCHUNK; ++ch) {
        const int q0 = ch * CHUNK;
        // ---- build the three X chunks in LDS with packed fp16 math ----
        for (int cp = lane; cp < CHUNK/2; cp += 32) {
            const int col = cp * 2;
            const int q = q0 + col;
            int jj, p;
            if (q < 1024) { jj = q >> 5;          p = q & 31; }
            else { const int qq = q - 1024; jj = 32 + qq / 48; p = 32 + qq % 48; }
            #pragma unroll
            for (int r = 0; r < 16; ++r) {
                const _Float16 hv = hs[r][jj];
                h2 hh; hh[0] = hv; hh[1] = hv;
                #pragma unroll
                for (int m = 0; m < 3; ++m) {
                    const h2 zv = *(const h2*)&zs[m][r][p];
                    *(h2*)&Xs[m][r][col] = hh * zv;
                }
            }
        }
        // ---- prefetch next chunk's B while VALU is busy (global_prefetch_b8)
        if (ch + 1 < NCHUNK) {
            #pragma unroll
            for (int nt = 0; nt < NT; ++nt) {
                const char* pb = (const char*)(Bpack +
                    ((size_t)(nt*KSTEPS + (ch + 1)*KSPC))*32*16);
                __builtin_prefetch(pb + lane*256,        0, 1);
                __builtin_prefetch(pb + lane*256 + 128,  0, 1);
            }
        }
        __syncthreads();

        for (int ks = 0; ks < KSPC; ++ks) {
            const int co = ks*32 + g*16;
            v16h A[3];
            #pragma unroll
            for (int m = 0; m < 3; ++m) {
                const _Float16* arow = &Xs[m][row][co];
                const v8h alo = *(const v8h*)(arow);
                const v8h ahi = *(const v8h*)(arow + 8);
                A[m] = __builtin_shufflevector(alo, ahi,
                    0,1,2,3,4,5,6,7,8,9,10,11,12,13,14,15);
            }
            const int gk = ch * KSPC + ks;
            #pragma unroll
            for (int nt = 0; nt < NT; ++nt) {
                const _Float16* bp = Bpack +
                    (((size_t)(nt*KSTEPS + gk))*32 + (size_t)lane)*16;
                const v8h blo = *(const v8h*)(bp);
                const v8h bhi = *(const v8h*)(bp + 8);
                const v16h B = __builtin_shufflevector(blo, bhi,
                    0,1,2,3,4,5,6,7,8,9,10,11,12,13,14,15);
                #pragma unroll
                for (int m = 0; m < 3; ++m)
                    acc[m][nt] = __builtin_amdgcn_wmma_f32_16x16x32_f16(
                        false, A[m], false, B, (short)0, acc[m][nt], false, false);
            }
        }
        __syncthreads();
    }

    // ---- scatter-sum into atoms[dst][k][m] ----
    const int kcol = lane & 15;
    #pragma unroll
    for (int nt = 0; nt < NT; ++nt) {
        const int k = nt*16 + kcol;
        if (k < COUTC) {
            #pragma unroll
            for (int m = 0; m < 3; ++m) {
                #pragma unroll
                for (int r = 0; r < 8; ++r) {
                    const int erow = r + g*8;
                    atomicAdd(&atoms[(size_t)dsts[erow]*273 + k*3 + m],
                              acc[m][nt][r]);
                }
            }
        }
    }
}

// ---------------------------------------------------------------------------
// Kernel D: self-interaction + NormSE3 + 3 residual refine blocks -> d_out
// ---------------------------------------------------------------------------
__device__ __forceinline__ void norm_se3_inplace(
    float* arr, int C, const float* w, const float* b, float* red, int t)
{
    float nk = 0.f, x0 = 0.f, x1 = 0.f, x2 = 0.f;
    if (t < C) {
        x0 = arr[t*3+0]; x1 = arr[t*3+1]; x2 = arr[t*3+2];
        nk = fmaxf(sqrtf(x0*x0 + x1*x1 + x2*x2), NORM_CLAMP);
    }
    red[t] = (t < C) ? nk : 0.f;
    __syncthreads();
    for (int s = 128; s > 0; s >>= 1) { if (t < s) red[t] += red[t+s]; __syncthreads(); }
    const float mu = red[0] / (float)C;
    __syncthreads();
    const float dn = nk - mu;
    red[t] = (t < C) ? dn*dn : 0.f;
    __syncthreads();
    for (int s = 128; s > 0; s >>= 1) { if (t < s) red[t] += red[t+s]; __syncthreads(); }
    const float var = red[0] / (float)C;
    __syncthreads();
    if (t < C) {
        const float nn = fmaxf(dn * rsqrtf(var + EPSV) * w[t] + b[t], 0.f);
        const float sc = nn / nk;
        arr[t*3+0] = x0 * sc; arr[t*3+1] = x1 * sc; arr[t*3+2] = x2 * sc;
    }
    __syncthreads();
}

__global__ __launch_bounds__(256) void node_post_kernel(
    const float* __restrict__ atoms, const float* __restrict__ t1f,
    const float* __restrict__ Wself,
    const float* __restrict__ gnw, const float* __restrict__ gnb,
    const float* __restrict__ refWa,
    const float* __restrict__ refgnw, const float* __restrict__ refgnb,
    const float* __restrict__ refWb,
    float* __restrict__ out)
{
    __shared__ float a[COUTC*3];
    __shared__ float y[182*3];
    __shared__ float red[256];
    __shared__ float t1s[48];
    const int n = blockIdx.x;
    const int t = threadIdx.x;

    if (t < 48) t1s[t] = t1f[n*48 + t];
    __syncthreads();

    if (t < COUTC) {
        #pragma unroll
        for (int m = 0; m < 3; ++m) {
            float v = atoms[(size_t)n*273 + t*3 + m];
            #pragma unroll
            for (int i = 0; i < 16; ++i) v += Wself[t*16 + i] * t1s[i*3 + m];
            a[t*3 + m] = v;
        }
    }
    __syncthreads();
    norm_se3_inplace(a, COUTC, gnw, gnb, red, t);

    for (int it = 0; it < 3; ++it) {
        const float* Wa = refWa + it*182*COUTC;
        const float* Wb = refWb + it*COUTC*182;
        if (t < 182) {
            float y0 = 0.f, y1 = 0.f, y2 = 0.f;
            for (int k = 0; k < COUTC; ++k) {
                const float w = Wa[t*COUTC + k];
                y0 += w * a[k*3+0]; y1 += w * a[k*3+1]; y2 += w * a[k*3+2];
            }
            y[t*3+0] = y0; y[t*3+1] = y1; y[t*3+2] = y2;
        }
        __syncthreads();
        norm_se3_inplace(y, 182, refgnw + it*182, refgnb + it*182, red, t);
        float yb0 = 0.f, yb1 = 0.f, yb2 = 0.f;
        if (t < COUTC) {
            for (int h = 0; h < 182; ++h) {
                const float w = Wb[t*182 + h];
                yb0 += w * y[h*3+0]; yb1 += w * y[h*3+1]; yb2 += w * y[h*3+2];
            }
            a[t*3+0] += yb0; a[t*3+1] += yb1; a[t*3+2] += yb2;
        }
        __syncthreads();
    }

    if (t < COUTC) {
        #pragma unroll
        for (int m = 0; m < 3; ++m) out[(size_t)n*273 + t*3 + m] = a[t*3 + m];
    }
}

// ---------------------------------------------------------------------------
extern "C" void kernel_launch(void* const* d_in, const int* in_sizes, int n_in,
                              void* d_out, int out_size, void* d_ws, size_t ws_size,
                              hipStream_t stream)
{
    (void)in_sizes; (void)n_in; (void)out_size; (void)ws_size;
    const float* density_0    = (const float*)d_in[0];
    const float* density_1    = (const float*)d_in[1];
    const float* seq_0        = (const float*)d_in[2];
    const float* edge_scalars = (const float*)d_in[3];
    const float* basis_01     = (const float*)d_in[4];
    const float* basis_11     = (const float*)d_in[5];
    const float* Wt0          = (const float*)d_in[6];
    const float* Wt1          = (const float*)d_in[7];
    const float* ln0_w        = (const float*)d_in[8];
    const float* ln0_b        = (const float*)d_in[9];
    const float* ln1_w        = (const float*)d_in[10];
    const float* ln1_b        = (const float*)d_in[11];
    const float* r01_W1       = (const float*)d_in[12];
    const float* r01_b1       = (const float*)d_in[13];
    const float* r01_W2       = (const float*)d_in[14];
    const float* r01_b2       = (const float*)d_in[15];
    const float* r01_W3       = (const float*)d_in[16];
    const float* r11_W1       = (const float*)d_in[17];
    const float* r11_b1       = (const float*)d_in[18];
    const float* r11_W2       = (const float*)d_in[19];
    const float* r11_b2       = (const float*)d_in[20];
    const float* r11_W3       = (const float*)d_in[21];
    const float* W_self       = (const float*)d_in[22];
    const float* gn_w         = (const float*)d_in[23];
    const float* gn_b         = (const float*)d_in[24];
    const float* ref_Wa       = (const float*)d_in[25];
    const float* ref_gn_w     = (const float*)d_in[26];
    const float* ref_gn_b     = (const float*)d_in[27];
    const float* ref_Wb       = (const float*)d_in[28];
    const int*   src          = (const int*)d_in[29];
    const int*   dst          = (const int*)d_in[30];

    char* ws = (char*)d_ws;
    float*     t0f   = (float*)(ws + 0);           // N*32 f32       =  512 KB
    float*     t1f   = (float*)(ws + 524288);      // N*48 f32       =  768 KB
    _Float16*  Hws   = (_Float16*)(ws + 1310720);  // E*64 f16       =    4 MB
    _Float16*  Zws   = (_Float16*)(ws + 5505024);  // E*240 f16      =   15 MB
    _Float16*  Bpack = (_Float16*)(ws + 21233664); // 6*80*32*16 f16 =  480 KB
    float*     atoms = (float*)(ws + 21725184);    // N*273 f32      = 4.47 MB

    hipMemsetAsync(atoms, 0, (size_t)NN * 273 * sizeof(float), stream);

    node_prep_kernel<<<NN, 64, 0, stream>>>(
        density_0, density_1, seq_0, Wt0, Wt1,
        ln0_w, ln0_b, ln1_w, ln1_b, t0f, t1f);

    pack_wbig_kernel<<<(NT*KSTEPS*32*16)/256, 256, 0, stream>>>(
        r01_W3, r11_W3, Bpack);

    edge_prep_kernel<<<EE, 32, 0, stream>>>(
        edge_scalars, basis_01, basis_11,
        r01_W1, r01_b1, r01_W2, r01_b2,
        r11_W1, r11_b1, r11_W2, r11_b2,
        src, t0f, t1f, Hws, Zws);

    conv_gemm_kernel<<<ETILES, 32, 0, stream>>>(
        Hws, Zws, Bpack, dst, atoms);

    node_post_kernel<<<NN, 256, 0, stream>>>(
        atoms, t1f, W_self, gn_w, gn_b,
        ref_Wa, ref_gn_w, ref_gn_b, ref_Wb, (float*)d_out);
}
